// MambaExpert_82145544503971
// MI455X (gfx1250) — matrix operand order repaired and checked by reading, not compiled, not run on hardware
//
#include <hip/hip_runtime.h>
#include <hip/hip_bf16.h>
#include <math.h>

// ---------------------------------------------------------------------------
// Mamba-style expert forward for MI455X (gfx1250, wave32, WMMA bf16).
//  K0  A_mean[j] = -(1/128) * sum_{i<128} exp(A_log[i,j])
//  K1  xp = emb[x] @ W_in + b_in              (bf16 WMMA, M=32768,N=1024,K=512, double-buffered)
//  K2  xs = silu(depthwise_conv3(xp)+conv_b)  (float4 vectorized, memory-bound)
//  K3  fused GEMM [softplus-dt-mean | x_state](bf16 WMMA, M=32768,N=128(x2),K=1024)
//  K4  parallel linear scan (chunked) -> sum_s h
//  K5  sum_s xs  (float4 chunks + atomics)
//  K6  pooled = tile(hsum)/S + (xs_sum/S)*D
//  K7  y = pooled @ W_out + b_out ; LayerNorm -> out
// ---------------------------------------------------------------------------

typedef __attribute__((ext_vector_type(16))) __bf16 v16bf;
typedef __attribute__((ext_vector_type(8)))  float  v8f;

#define D_MODEL 512
#define D_INNER 1024
#define NSTATE  128
#define BATCH   16
#define SEQ     2048
#define MTOT    (BATCH * SEQ)       // 32768 rows
#define LDSW    40                  // LDS row stride in bf16 (80B, multiple of 16B)

static __device__ __forceinline__ float softplusf(float z) {
    return (z > 20.0f) ? z : log1pf(expf(z));
}
static __device__ __forceinline__ float siluf(float z) {
    return z / (1.0f + expf(-z));
}
static __device__ __forceinline__ unsigned int pack_bf16(float a, float b) {
    union { __bf16 h[2]; unsigned int u; } c;
    c.h[0] = (__bf16)a; c.h[1] = (__bf16)b;
    return c.u;
}

// ---------------------------------------------------------------------------
// K0
// ---------------------------------------------------------------------------
__global__ void k0_amean(const float* __restrict__ A_log, float* __restrict__ A_mean) {
    int j = threadIdx.x;           // 0..127
    float acc = 0.0f;
    for (int i = 0; i < NSTATE; ++i) acc += expf(A_log[i * NSTATE + j]);
    A_mean[j] = -acc * (1.0f / 128.0f);
}

// ---------------------------------------------------------------------------
// K1: fused embedding gather + GEMM, 128x128 tile, K-chunks of 32,
// double-buffered LDS, packed bf16 staging.
// ---------------------------------------------------------------------------
__global__ __launch_bounds__(256)
void k1_embed_gemm(const int* __restrict__ tokens,
                   const float* __restrict__ emb,
                   const float* __restrict__ W_in,
                   const float* __restrict__ b_in,
                   float* __restrict__ xp) {
    __shared__ __align__(16) __bf16 Asm[2][128 * LDSW];
    __shared__ __align__(16) __bf16 Bsm[2][128 * LDSW];
    __shared__ int tok[128];

    const int tid   = threadIdx.x;
    const int mBase = blockIdx.x * 128;
    const int nBase = blockIdx.y * 128;
    const int w     = tid >> 5;
    const int lane  = tid & 31;
    const int lo    = lane & 15;
    const int hi    = lane >> 4;

    if (tid < 128) tok[tid] = tokens[mBase + tid];
    __syncthreads();

    float2 ar[8];
    float4 br[4];

    auto loadA = [&](int kBase) {
#pragma unroll
        for (int j = 0; j < 8; ++j) {
            int p = j * 256 + tid;          // pair index 0..2047
            int r = p >> 4, kp = p & 15;
            ar[j] = *(const float2*)(emb + (size_t)tok[r] * D_MODEL + kBase + kp * 2);
        }
    };
    auto loadB = [&](int kBase) {
#pragma unroll
        for (int j = 0; j < 4; ++j) {
            int g = j * 256 + tid;          // group of 4 n, 0..1023
            int k = g >> 5, n = (g & 31) * 4;
            br[j] = *(const float4*)(W_in + (size_t)(kBase + k) * D_INNER + nBase + n);
        }
    };
    auto storeA = [&](int buf) {
        unsigned int* a32 = (unsigned int*)&Asm[buf][0];
#pragma unroll
        for (int j = 0; j < 8; ++j) {
            int p = j * 256 + tid;
            int r = p >> 4, kp = p & 15;
            a32[r * (LDSW / 2) + kp] = pack_bf16(ar[j].x, ar[j].y);
        }
    };
    auto storeB = [&](int buf) {
#pragma unroll
        for (int j = 0; j < 4; ++j) {
            int g = j * 256 + tid;
            int k = g >> 5, n = (g & 31) * 4;
            Bsm[buf][(n + 0) * LDSW + k] = (__bf16)br[j].x;
            Bsm[buf][(n + 1) * LDSW + k] = (__bf16)br[j].y;
            Bsm[buf][(n + 2) * LDSW + k] = (__bf16)br[j].z;
            Bsm[buf][(n + 3) * LDSW + k] = (__bf16)br[j].w;
        }
    };

    v8f acc[8];
#pragma unroll
    for (int nt = 0; nt < 8; ++nt) acc[nt] = (v8f){0,0,0,0,0,0,0,0};

    const int NIT = D_MODEL / 32;   // 16
    loadA(0); loadB(0);
    storeA(0); storeB(0);

    for (int it = 0; it < NIT; ++it) {
        __syncthreads();
        if (it + 1 < NIT) { loadA((it + 1) * 32); loadB((it + 1) * 32); }

        const int buf  = it & 1;
        v16bf av;
        const int arow = (w * 16 + lo) * LDSW;
        const int kb   = hi * 16;
#pragma unroll
        for (int i = 0; i < 8; ++i) {
            av[i]     = Asm[buf][arow + kb + i];
            av[8 + i] = Asm[buf][arow + kb + 8 + i];
        }
#pragma unroll
        for (int nt = 0; nt < 8; ++nt) {
            v16bf bv;
            const int brow = (nt * 16 + lo) * LDSW;
            const int kbb  = hi * 16;
#pragma unroll
            for (int i = 0; i < 16; ++i) bv[i] = Bsm[buf][brow + kbb + i];
            acc[nt] = __builtin_amdgcn_wmma_f32_16x16x32_bf16(
                false, av, false, bv, (short)0, acc[nt], false, false);
        }
        if (it + 1 < NIT) { storeA(1 - buf); storeB(1 - buf); }
    }

#pragma unroll
    for (int nt = 0; nt < 8; ++nt) {
        const int col = nBase + nt * 16 + lo;
        const float bias = b_in[col];
#pragma unroll
        for (int v = 0; v < 8; ++v) {
            const int row = mBase + w * 16 + v + 8 * hi;
            xp[(size_t)row * D_INNER + col] = acc[nt][v] + bias;
        }
    }
}

// ---------------------------------------------------------------------------
// K2: depthwise conv (k=3, pad 1 along S) + bias + SiLU, float4 over channels
// ---------------------------------------------------------------------------
__global__ void k2_conv_silu(const float* __restrict__ xp,
                             const float* __restrict__ conv_w,
                             const float* __restrict__ conv_b,
                             float* __restrict__ xs) {
    const int idx = blockIdx.x * blockDim.x + threadIdx.x;  // 0..8388607
    const int c   = (idx & 255) * 4;
    const int s   = (idx >> 8) & (SEQ - 1);
    const size_t base = (size_t)idx * 4;

    float4 x0 = *(const float4*)(xp + base);
    float4 xm = {0,0,0,0}, xq = {0,0,0,0};
    if (s > 0)       xm = *(const float4*)(xp + base - D_INNER);
    if (s < SEQ - 1) xq = *(const float4*)(xp + base + D_INNER);
    float4 cb = *(const float4*)(conv_b + c);

    const float a0[4] = {xm.x, xm.y, xm.z, xm.w};
    const float a1[4] = {x0.x, x0.y, x0.z, x0.w};
    const float a2[4] = {xq.x, xq.y, xq.z, xq.w};
    const float bb[4] = {cb.x, cb.y, cb.z, cb.w};
    float4 out;
    float r[4];
#pragma unroll
    for (int i = 0; i < 4; ++i) {
        float v = bb[i]
                + a0[i] * conv_w[(c + i) * 3 + 0]
                + a1[i] * conv_w[(c + i) * 3 + 1]
                + a2[i] * conv_w[(c + i) * 3 + 2];
        r[i] = siluf(v);
    }
    out.x = r[0]; out.y = r[1]; out.z = r[2]; out.w = r[3];
    *(float4*)(xs + base) = out;
}

// ---------------------------------------------------------------------------
// K3: fused GEMM on xs. blockIdx.y==0 -> dt half (softplus + row-mean),
//     blockIdx.y==1 -> x_state half (+ b_x). Double-buffered as K1.
// ---------------------------------------------------------------------------
__global__ __launch_bounds__(256)
void k3_dt_state_gemm(const float* __restrict__ xs,
                      const float* __restrict__ W_dt,
                      const float* __restrict__ b_dt,
                      const float* __restrict__ W_x,
                      const float* __restrict__ b_x,
                      float* __restrict__ s_dt,
                      float* __restrict__ x_state) {
    __shared__ __align__(16) __bf16 Asm[2][128 * LDSW];
    __shared__ __align__(16) __bf16 Bsm[2][128 * LDSW];
    __shared__ float red[256 * 8];

    const int tid   = threadIdx.x;
    const int mBase = blockIdx.x * 128;
    const int half  = blockIdx.y;          // 0 = dt, 1 = x_state
    const int w     = tid >> 5;
    const int lane  = tid & 31;
    const int lo    = lane & 15;
    const int hi    = lane >> 4;

    const float* Bsrc = half ? W_x : W_dt;
    const int    ldB  = half ? NSTATE : D_INNER;

    float2 ar[8];
    float4 br[4];

    auto loadA = [&](int kBase) {
#pragma unroll
        for (int j = 0; j < 8; ++j) {
            int p = j * 256 + tid;
            int r = p >> 4, kp = p & 15;
            ar[j] = *(const float2*)(xs + (size_t)(mBase + r) * D_INNER + kBase + kp * 2);
        }
    };
    auto loadB = [&](int kBase) {
#pragma unroll
        for (int j = 0; j < 4; ++j) {
            int g = j * 256 + tid;
            int k = g >> 5, n = (g & 31) * 4;
            br[j] = *(const float4*)(Bsrc + (size_t)(kBase + k) * ldB + n);
        }
    };
    auto storeA = [&](int buf) {
        unsigned int* a32 = (unsigned int*)&Asm[buf][0];
#pragma unroll
        for (int j = 0; j < 8; ++j) {
            int p = j * 256 + tid;
            int r = p >> 4, kp = p & 15;
            a32[r * (LDSW / 2) + kp] = pack_bf16(ar[j].x, ar[j].y);
        }
    };
    auto storeB = [&](int buf) {
#pragma unroll
        for (int j = 0; j < 4; ++j) {
            int g = j * 256 + tid;
            int k = g >> 5, n = (g & 31) * 4;
            Bsm[buf][(n + 0) * LDSW + k] = (__bf16)br[j].x;
            Bsm[buf][(n + 1) * LDSW + k] = (__bf16)br[j].y;
            Bsm[buf][(n + 2) * LDSW + k] = (__bf16)br[j].z;
            Bsm[buf][(n + 3) * LDSW + k] = (__bf16)br[j].w;
        }
    };

    v8f acc[8];
#pragma unroll
    for (int nt = 0; nt < 8; ++nt) acc[nt] = (v8f){0,0,0,0,0,0,0,0};

    const int NIT = D_INNER / 32;  // 32
    loadA(0); loadB(0);
    storeA(0); storeB(0);

    for (int it = 0; it < NIT; ++it) {
        __syncthreads();
        if (it + 1 < NIT) { loadA((it + 1) * 32); loadB((it + 1) * 32); }

        const int buf  = it & 1;
        v16bf av;
        const int arow = (w * 16 + lo) * LDSW;
        const int kb   = hi * 16;
#pragma unroll
        for (int i = 0; i < 8; ++i) {
            av[i]     = Asm[buf][arow + kb + i];
            av[8 + i] = Asm[buf][arow + kb + 8 + i];
        }
#pragma unroll
        for (int nt = 0; nt < 8; ++nt) {
            v16bf bv;
            const int brow = (nt * 16 + lo) * LDSW;
            const int kbb  = hi * 16;
#pragma unroll
            for (int i = 0; i < 16; ++i) bv[i] = Bsm[buf][brow + kbb + i];
            acc[nt] = __builtin_amdgcn_wmma_f32_16x16x32_bf16(
                false, av, false, bv, (short)0, acc[nt], false, false);
        }
        if (it + 1 < NIT) { storeA(1 - buf); storeB(1 - buf); }
    }

    if (half == 1) {
#pragma unroll
        for (int nt = 0; nt < 8; ++nt) {
            const int col = nt * 16 + lo;
            const float bias = b_x[col];
#pragma unroll
            for (int v = 0; v < 8; ++v) {
                const int row = mBase + w * 16 + v + 8 * hi;
                x_state[(size_t)row * NSTATE + col] = acc[nt][v] + bias;
            }
        }
    } else {
        float part[8];
#pragma unroll
        for (int v = 0; v < 8; ++v) part[v] = 0.0f;
#pragma unroll
        for (int nt = 0; nt < 8; ++nt) {
            const float bias = b_dt[nt * 16 + lo];
#pragma unroll
            for (int v = 0; v < 8; ++v) part[v] += softplusf(acc[nt][v] + bias);
        }
#pragma unroll
        for (int v = 0; v < 8; ++v) red[tid * 8 + v] = part[v];
        __syncthreads();
        if (tid < 128) {
            const int row_l = tid;
            const int ww = row_l >> 4;
            const int rem = row_l & 15;
            const int hh = rem >> 3;
            const int vv = rem & 7;
            float sum = 0.0f;
#pragma unroll
            for (int l = 0; l < 16; ++l) sum += red[(ww * 32 + hh * 16 + l) * 8 + vv];
            s_dt[mBase + row_l] = sum * (1.0f / 128.0f);
        }
    }
}

// ---------------------------------------------------------------------------
// K4a: per-chunk scan summaries. chain=(b,n), 16 chunks of 128 steps.
// ---------------------------------------------------------------------------
__global__ void k4a_scan_chunks(const float* __restrict__ s_dt,
                                const float* __restrict__ x_state,
                                const float* __restrict__ A_mean,
                                float* __restrict__ P, float* __restrict__ SQ,
                                float* __restrict__ HE, float* __restrict__ HS) {
    const int t  = blockIdx.x * blockDim.x + threadIdx.x;   // 0..32767
    const int n  = t & 127;
    const int ch = (t >> 7) & 15;
    const int b  = t >> 11;
    const float am = A_mean[n];

    float Q = 1.0f, sq = 0.0f, h = 0.0f, hs = 0.0f;
    const int s0 = ch * 128;
    for (int j = 0; j < 128; ++j) {
        const int s = s0 + j;
        const float a  = expf(am * s_dt[b * SEQ + s]);
        const float xv = x_state[(size_t)(b * SEQ + s) * NSTATE + n];
        h = h * a + xv;
        hs += h;
        Q *= a;
        sq += Q;
    }
    const int idx = (b * 16 + ch) * NSTATE + n;
    P[idx] = Q; SQ[idx] = sq; HE[idx] = h; HS[idx] = hs;
}

// ---------------------------------------------------------------------------
// K4b: serial combine of 16 chunks per chain
// ---------------------------------------------------------------------------
__global__ void k4b_scan_combine(const float* __restrict__ P, const float* __restrict__ SQ,
                                 const float* __restrict__ HE, const float* __restrict__ HS,
                                 float* __restrict__ hsum) {
    const int t = blockIdx.x * blockDim.x + threadIdx.x;    // 0..2047
    const int n = t & 127;
    const int b = t >> 7;
    float h = 0.0f, total = 0.0f;
    for (int ch = 0; ch < 16; ++ch) {
        const int idx = (b * 16 + ch) * NSTATE + n;
        total += h * SQ[idx] + HS[idx];
        h = h * P[idx] + HE[idx];
    }
    hsum[b * NSTATE + n] = total;
}

// ---------------------------------------------------------------------------
// K5: xs_sum[b][c] = sum_s xs[b,s,c]  (float4, 8 S-chunks + atomics)
// ---------------------------------------------------------------------------
__global__ void k5_xs_sum(const float* __restrict__ xs, float* __restrict__ xs_sum) {
    const int t  = blockIdx.x * blockDim.x + threadIdx.x;   // 0..32767
    const int c  = (t & 255) * 4;
    const int ch = (t >> 8) & 7;
    const int b  = t >> 11;
    float4 sum = {0,0,0,0};
    const int s0 = ch * 256;
    for (int j = 0; j < 256; ++j) {
        float4 v = *(const float4*)(xs + (size_t)(b * SEQ + s0 + j) * D_INNER + c);
        sum.x += v.x; sum.y += v.y; sum.z += v.z; sum.w += v.w;
    }
    atomicAdd(&xs_sum[b * D_INNER + c + 0], sum.x);
    atomicAdd(&xs_sum[b * D_INNER + c + 1], sum.y);
    atomicAdd(&xs_sum[b * D_INNER + c + 2], sum.z);
    atomicAdd(&xs_sum[b * D_INNER + c + 3], sum.w);
}

// ---------------------------------------------------------------------------
// K6: pooled
// ---------------------------------------------------------------------------
__global__ void k6_pooled(const float* __restrict__ hsum, const float* __restrict__ xs_sum,
                          const float* __restrict__ Dvec, float* __restrict__ pooled) {
    const int b = blockIdx.x;
    const int c = threadIdx.x;
    const float invS = 1.0f / (float)SEQ;
    pooled[b * D_INNER + c] =
        hsum[b * NSTATE + (c & (NSTATE - 1))] * invS +
        xs_sum[b * D_INNER + c] * invS * Dvec[c];
}

// ---------------------------------------------------------------------------
// K7: y = pooled @ W_out + b_out ; LayerNorm(512) -> out
// ---------------------------------------------------------------------------
__global__ __launch_bounds__(512)
void k7_out_ln(const float* __restrict__ pooled, const float* __restrict__ W_out,
               const float* __restrict__ b_out, const float* __restrict__ ln_g,
               const float* __restrict__ ln_b, float* __restrict__ out) {
    __shared__ float pl[D_INNER];
    __shared__ float rs[512];
    __shared__ float rq[512];
    const int b = blockIdx.x;
    const int o = threadIdx.x;

    pl[o]       = pooled[b * D_INNER + o];
    pl[o + 512] = pooled[b * D_INNER + o + 512];
    __syncthreads();

    float acc = b_out[o];
    for (int c = 0; c < D_INNER; ++c)
        acc += pl[c] * W_out[(size_t)c * D_MODEL + o];

    rs[o] = acc;
    rq[o] = acc * acc;
    __syncthreads();
    for (int st = 256; st > 0; st >>= 1) {
        if (o < st) { rs[o] += rs[o + st]; rq[o] += rq[o + st]; }
        __syncthreads();
    }
    const float mu  = rs[0] * (1.0f / 512.0f);
    const float var = rq[0] * (1.0f / 512.0f) - mu * mu;
    out[b * D_MODEL + o] = (acc - mu) * rsqrtf(var + 1e-5f) * ln_g[o] + ln_b[o];
}

// ---------------------------------------------------------------------------
// Host launcher
// ---------------------------------------------------------------------------
extern "C" void kernel_launch(void* const* d_in, const int* in_sizes, int n_in,
                              void* d_out, int out_size, void* d_ws, size_t ws_size,
                              hipStream_t stream) {
    (void)in_sizes; (void)n_in; (void)out_size; (void)ws_size;

    const int*   x      = (const int*)  d_in[0];
    const float* emb    = (const float*)d_in[1];
    const float* W_in   = (const float*)d_in[2];
    const float* b_in   = (const float*)d_in[3];
    const float* conv_w = (const float*)d_in[4];
    const float* conv_b = (const float*)d_in[5];
    const float* W_x    = (const float*)d_in[6];
    const float* b_x    = (const float*)d_in[7];
    const float* W_dt   = (const float*)d_in[8];
    const float* b_dt   = (const float*)d_in[9];
    const float* A_log  = (const float*)d_in[10];
    const float* Dvec   = (const float*)d_in[11];
    const float* W_out  = (const float*)d_in[12];
    const float* b_out  = (const float*)d_in[13];
    const float* ln_g   = (const float*)d_in[14];
    const float* ln_b   = (const float*)d_in[15];
    float* out = (float*)d_out;

    // workspace layout (floats)
    float* ws = (float*)d_ws;
    const size_t XP    = 0;
    const size_t XS    = XP + (size_t)MTOT * D_INNER;
    const size_t XST   = XS + (size_t)MTOT * D_INNER;
    const size_t SDT   = XST + (size_t)MTOT * NSTATE;
    const size_t AMEAN = SDT + (size_t)MTOT;
    const size_t CP    = AMEAN + NSTATE;
    const size_t CSQ   = CP  + (size_t)BATCH * 16 * NSTATE;
    const size_t CHE   = CSQ + (size_t)BATCH * 16 * NSTATE;
    const size_t CHS   = CHE + (size_t)BATCH * 16 * NSTATE;
    const size_t HSUM  = CHS + (size_t)BATCH * 16 * NSTATE;
    const size_t XSSUM = HSUM + (size_t)BATCH * NSTATE;
    const size_t POOL  = XSSUM + (size_t)BATCH * D_INNER;

    hipMemsetAsync(ws + XSSUM, 0, (size_t)BATCH * D_INNER * sizeof(float), stream);

    k0_amean<<<1, 128, 0, stream>>>(A_log, ws + AMEAN);

    k1_embed_gemm<<<dim3(MTOT / 128, D_INNER / 128), 256, 0, stream>>>(
        x, emb, W_in, b_in, ws + XP);

    k2_conv_silu<<<(MTOT * D_INNER / 4) / 256, 256, 0, stream>>>(
        ws + XP, conv_w, conv_b, ws + XS);

    k3_dt_state_gemm<<<dim3(MTOT / 128, 2), 256, 0, stream>>>(
        ws + XS, W_dt, b_dt, W_x, b_x, ws + SDT, ws + XST);

    k4a_scan_chunks<<<(BATCH * 16 * NSTATE) / 256, 256, 0, stream>>>(
        ws + SDT, ws + XST, ws + AMEAN, ws + CP, ws + CSQ, ws + CHE, ws + CHS);

    k4b_scan_combine<<<(BATCH * NSTATE) / 256, 256, 0, stream>>>(
        ws + CP, ws + CSQ, ws + CHE, ws + CHS, ws + HSUM);

    k5_xs_sum<<<(BATCH * 8 * D_INNER / 4) / 256, 256, 0, stream>>>(ws + XS, ws + XSSUM);

    k6_pooled<<<BATCH, D_INNER, 0, stream>>>(ws + HSUM, ws + XSSUM, Dvec, ws + POOL);

    k7_out_ln<<<BATCH, D_MODEL, 0, stream>>>(ws + POOL, W_out, b_out, ln_g, ln_b, out);
}
